// Decoder_78761110274054
// MI455X (gfx1250) — matrix operand order repaired and checked by reading, not compiled, not run on hardware
//
#include <hip/hip_runtime.h>
#include <hip/hip_bf16.h>
#include <stdint.h>

typedef __attribute__((ext_vector_type(16))) __bf16 v16bf;
typedef __attribute__((ext_vector_type(8)))  float  v8f;
typedef __attribute__((ext_vector_type(4)))  unsigned int v4u;
typedef __attribute__((ext_vector_type(8)))  int    v8i;
typedef __attribute__((ext_vector_type(4)))  int    v4i;

#define B_    16
#define TENC  256
#define TDEC  200
#define NMEL  80
#define EDIM  512
#define HDIM  1024
#define ADIM  128
#define PDIM  256
#define FDIM  32
#define KCONV 31
#define NGATE 4096
#define XK_ATT 1792   // [px(256) | ctx(512) | h1(1024)]
#define XK_DEC 2560   // [h1(1024) | ctx(512) | h2(1024)]

#if defined(__HIP_DEVICE_COMPILE__) && defined(__has_builtin)
#  if __has_builtin(__builtin_amdgcn_tensor_load_to_lds)
#    define HAVE_TDM 1
#  else
#    define HAVE_TDM 0
#  endif
#else
#  define HAVE_TDM 0
#endif

__device__ __forceinline__ uint32_t bf16_rne(float f) {
  uint32_t u = __float_as_uint(f);
  return (u + 0x7FFFu + ((u >> 16) & 1u)) >> 16;
}
__device__ __forceinline__ float sigmoidf_(float x) {
  return 1.0f / (1.0f + __expf(-x));
}
// generic LDS pointer -> hardware LDS byte offset (low 32 bits of flat address)
__device__ __forceinline__ uint32_t lds_off32(const void* p) {
  return (uint32_t)(uintptr_t)p;
}

// ---------------------------------------------------------------------------
// Repack f32 weights [Wih; Whh] (row-major K x 4096) into fragment-major bf16.
// B-fragment k order used here (16-bit B, 32x16 tile, wave32; contiguous k, A reads use the same):
//   lane L: n = L&15, kbase = (L>>4)*16; VGPR j holds {B[kbase+2j,n], B[kbase+2j+1,n]}
// Stored so each lane's 8 dwords are contiguous: idx = ((nt*KT+kt)*32+lane)*8 + j
// ---------------------------------------------------------------------------
__global__ void convert_w_kernel(const float* __restrict__ Wih,
                                 const float* __restrict__ Whh,
                                 int K1, int KT, uint32_t* __restrict__ out) {
  size_t total = (size_t)256 * KT * 256;
  size_t stride = (size_t)gridDim.x * blockDim.x;
  for (size_t idx = (size_t)blockIdx.x * blockDim.x + threadIdx.x; idx < total; idx += stride) {
    int j    = (int)(idx & 7);
    int lane = (int)((idx >> 3) & 31);
    size_t rest = idx >> 8;          // nt*KT + kt
    int kt = (int)(rest % KT);
    int nt = (int)(rest / KT);
    int k = kt * 32 + ((lane >> 4) << 4) + 2 * j;
    int n = nt * 16 + (lane & 15);
    float f0 = (k     < K1) ? Wih[(size_t)k * NGATE + n]       : Whh[(size_t)(k - K1) * NGATE + n];
    float f1 = (k + 1 < K1) ? Wih[(size_t)(k + 1) * NGATE + n] : Whh[(size_t)(k + 1 - K1) * NGATE + n];
    out[idx] = bf16_rne(f0) | (bf16_rne(f1) << 16);
  }
}

// ---------------------------------------------------------------------------
// PreNet: two Linear+ReLU (dropout = identity in eval)
// ---------------------------------------------------------------------------
__global__ void prenet1_kernel(const float* __restrict__ mel,
                               const float* __restrict__ W1, const float* __restrict__ b1,
                               float* __restrict__ tmp1) {
  size_t total = (size_t)B_ * TDEC * PDIM;
  size_t stride = (size_t)gridDim.x * blockDim.x;
  for (size_t idx = (size_t)blockIdx.x * blockDim.x + threadIdx.x; idx < total; idx += stride) {
    int c = (int)(idx & (PDIM - 1));
    int r = (int)(idx >> 8);
    int b = r / TDEC, t = r % TDEC;
    float s = b1[c];
    if (t > 0) {
      const float* x = mel + ((size_t)b * TDEC + (t - 1)) * NMEL;
      for (int k = 0; k < NMEL; ++k) s += x[k] * W1[k * PDIM + c];
    }
    tmp1[idx] = fmaxf(s, 0.0f);
  }
}
__global__ void prenet2_kernel(const float* __restrict__ tmp1,
                               const float* __restrict__ W2, const float* __restrict__ b2,
                               __hip_bfloat16* __restrict__ pre_b) {
  size_t total = (size_t)B_ * TDEC * PDIM;
  size_t stride = (size_t)gridDim.x * blockDim.x;
  for (size_t idx = (size_t)blockIdx.x * blockDim.x + threadIdx.x; idx < total; idx += stride) {
    int c = (int)(idx & (PDIM - 1));
    size_t r = idx >> 8;
    const float* x = tmp1 + r * PDIM;
    float s = b2[c];
    for (int k = 0; k < PDIM; ++k) s += x[k] * W2[k * PDIM + c];
    pre_b[idx] = __float2bfloat16(fmaxf(s, 0.0f));
  }
}

// keys[b,t,a] = enc[b,t,:] @ v_proj[:,a]
__global__ void keys_kernel(const float* __restrict__ enc, const float* __restrict__ v_proj,
                            float* __restrict__ keys) {
  size_t total = (size_t)B_ * TENC * ADIM;
  size_t stride = (size_t)gridDim.x * blockDim.x;
  for (size_t idx = (size_t)blockIdx.x * blockDim.x + threadIdx.x; idx < total; idx += stride) {
    int a = (int)(idx & (ADIM - 1));
    size_t bt = idx >> 7;
    const float* x = enc + bt * EDIM;
    float s = 0.0f;
    for (int e = 0; e < EDIM; ++e) s += x[e] * v_proj[e * ADIM + a];
    keys[idx] = s;
  }
}

__global__ void init_state_kernel(float* h1, float* c1, float* h2, float* c2,
                                  float* aw, float* caw, float* ctx,
                                  __hip_bfloat16* x_att, __hip_bfloat16* x_dec,
                                  const __hip_bfloat16* __restrict__ pre_b) {
  size_t stride = (size_t)gridDim.x * blockDim.x;
  size_t i0 = (size_t)blockIdx.x * blockDim.x + threadIdx.x;
  for (size_t i = i0; i < (size_t)B_ * HDIM; i += stride) { h1[i] = 0; c1[i] = 0; h2[i] = 0; c2[i] = 0; }
  for (size_t i = i0; i < (size_t)B_ * TENC; i += stride) { aw[i] = 0; caw[i] = 0; }
  for (size_t i = i0; i < (size_t)B_ * EDIM; i += stride) ctx[i] = 0;
  __hip_bfloat16 z = __float2bfloat16(0.0f);
  for (size_t i = i0; i < (size_t)B_ * XK_ATT; i += stride) {
    int b = (int)(i / XK_ATT), k = (int)(i % XK_ATT);
    x_att[i] = (k < PDIM) ? pre_b[((size_t)b * TDEC) * PDIM + k] : z;   // px_0, ctx=0, h1=0
  }
  for (size_t i = i0; i < (size_t)B_ * XK_DEC; i += stride) x_dec[i] = z;
}

// ---------------------------------------------------------------------------
// Fused LSTM step. Whole x (16 x xK bf16) staged into LDS once (TDM tensor
// DMA when available, else global_load_async_to_lds), with 4-dword padding
// every 256 dwords (lds_dw(e) = e + 4*(e>>8)) for 16B alignment +
// conflict-free ds_load_b128 A-fragment reads. Barrier-free WMMA K loop.
// Grid: 32 blocks x 256 threads (8 waves); wave owns one 16-col tile / gate.
// ---------------------------------------------------------------------------
__global__ void __launch_bounds__(256)
lstm_step_kernel(const uint32_t* __restrict__ Wfrag, const float* __restrict__ bias,
                 const uint32_t* __restrict__ xq, int xK,
                 float* __restrict__ c_state, float* __restrict__ h_out) {
  extern __shared__ uint32_t smem[];
  const int tid  = threadIdx.x;
  const int lane = tid & 31;
  const int wave = tid >> 5;
  const int hslice0 = blockIdx.x * 128;
  const int KT = xK >> 5;
  const int DW = xK >> 1;              // dwords per row of x
  const int ndw = DW * 16;             // total x dwords
  int xdw = ndw + 4 * (ndw >> 8) + 16; // padded LDS dwords for x (+slack)
  xdw = (xdw + 3) & ~3;
  uint32_t* xl = smem;
  float* gates = (float*)(smem + xdw); // 16 x 512 block-local gate cols

  // ---- stage x into LDS ----
#if HAVE_TDM
  if (tid == 0) {
    uint64_t ga = (uint64_t)(uintptr_t)xq;
    v4u g0;
    g0[0] = 1u;                                   // count=1, user descriptor
    g0[1] = lds_off32(xl);                        // lds_addr (bytes)
    g0[2] = (uint32_t)ga;                         // global_addr[31:0]
    g0[3] = (uint32_t)(ga >> 32) | (2u << 30);    // global_addr[56:32] | type=2
    v8i g1 = {};
    g1[0] = (int)((2u << 16)      // data_size = 4B
                | (1u << 20)      // pad_enable
                | (7u << 22)      // pad_interval = 256 dwords
                | (3u << 25));    // pad_amount  = 4 dwords
    g1[1] = (int)(((uint32_t)DW & 0xFFFFu) << 16);  // tensor_dim0[15:0] @ bits 63:48
    g1[2] = (int)(16u << 16);                       // tensor_dim1=16   @ bits 111:80
    g1[3] = (int)(((uint32_t)DW & 0xFFFFu) << 16);  // tile_dim0        @ bits 127:112
    g1[4] = 16;                                     // tile_dim1=16
    g1[5] = DW;                                     // tensor_dim0_stride[31:0]
    g1[6] = (int)(((uint32_t)DW & 0xFFFFu) << 16);  // tensor_dim1_stride low bits
    g1[7] = 0;
    v4i g2 = {}; v4i g3 = {};
#if defined(__clang_major__) && (__clang_major__ >= 23)
    v8i g4 = {};
    __builtin_amdgcn_tensor_load_to_lds(g0, g1, g2, g3, g4, 0);
#else
    __builtin_amdgcn_tensor_load_to_lds(g0, g1, g2, g3, 0);
#endif
#if __has_builtin(__builtin_amdgcn_s_wait_tensorcnt)
    __builtin_amdgcn_s_wait_tensorcnt(0);
#else
    asm volatile("s_wait_tensorcnt 0x0" ::: "memory");
#endif
  }
  __syncthreads();
#else
  {
    int nch = ndw >> 2;                         // 128-bit chunks
    for (int i = tid; i < nch; i += 256) {
      int e = i << 2;
      uint32_t loff = lds_off32(xl + e + 4 * (e >> 8));
      unsigned long long ga = (unsigned long long)(uintptr_t)(xq + e);
      asm volatile("global_load_async_to_lds_b128 %0, %1, off"
                   :: "v"(loff), "v"(ga) : "memory");
    }
    asm volatile("s_wait_asynccnt 0x0" ::: "memory");
    __syncthreads();
  }
#endif

  // ---- WMMA K loop (barrier-free) ----
  int nt[4];
  #pragma unroll
  for (int gi = 0; gi < 4; ++gi) nt[gi] = gi * 64 + blockIdx.x * 8 + wave;

  v8f acc[4] = {};
  const int m   = lane & 15;
  const int sel = lane >> 4;
  const int e_row = m * DW;

  for (int kt = 0; kt < KT; ++kt) {
    int e0 = e_row + kt * 16 + sel * 8;
    int e1 = e0 + 4;
    union { uint4 q[2]; v16bf v; } af;
    af.q[0] = *(const uint4*)(xl + e0 + 4 * (e0 >> 8));   // K = sel*16 .. +7
    af.q[1] = *(const uint4*)(xl + e1 + 4 * (e1 >> 8));   // K = sel*16+8 .. +15

    #pragma unroll
    for (int gi = 0; gi < 4; ++gi) {
      const uint32_t* bptr = Wfrag + (((size_t)nt[gi] * KT + kt) * 32 + lane) * 8;
      union { uint4 q[2]; v16bf v; } bf_;
      bf_.q[0] = *(const uint4*)(bptr);
      bf_.q[1] = *(const uint4*)(bptr + 4);
      acc[gi] = __builtin_amdgcn_wmma_f32_16x16x32_bf16(
          false, af.v, false, bf_.v, (short)0, acc[gi], false, false);
    }
  }

  // D layout: VGPR r, lanes 0-15 -> M=r, lanes 16-31 -> M=8+r; n = lane&15
  const int nloc = lane & 15, mbase = (lane >> 4) * 8;
  #pragma unroll
  for (int gi = 0; gi < 4; ++gi)
    #pragma unroll
    for (int r = 0; r < 8; ++r)
      gates[(mbase + r) * 512 + gi * 128 + wave * 16 + nloc] = acc[gi][r];
  __syncthreads();

  for (int idx = tid; idx < 16 * 128; idx += 256) {
    int mm = idx >> 7, hh = idx & 127;
    int h = hslice0 + hh;
    float gi_ = gates[mm * 512 + hh]       + bias[h];
    float gf  = gates[mm * 512 + 128 + hh] + bias[HDIM + h];
    float gg  = gates[mm * 512 + 256 + hh] + bias[2 * HDIM + h];
    float go  = gates[mm * 512 + 384 + hh] + bias[3 * HDIM + h];
    float c = sigmoidf_(gf) * c_state[mm * HDIM + h] + sigmoidf_(gi_) * tanhf(gg);
    c_state[mm * HDIM + h] = c;
    h_out[mm * HDIM + h]   = sigmoidf_(go) * tanhf(c);
  }
}

// ---------------------------------------------------------------------------
// Location-sensitive attention; one block per batch row. aw/caw staged into
// LDS with async global->LDS loads. Propagates h1/ctx bf16 into packed x.
// ---------------------------------------------------------------------------
__global__ void __launch_bounds__(256)
attention_kernel(const float* __restrict__ enc, const float* __restrict__ keys,
                 const float* __restrict__ h1, const float* __restrict__ q_proj,
                 const float* __restrict__ loc_conv, const float* __restrict__ loc_proj,
                 const float* __restrict__ score_b, const float* __restrict__ score_v,
                 float* __restrict__ aw, float* __restrict__ caw, float* __restrict__ ctx,
                 __hip_bfloat16* __restrict__ x_att, __hip_bfloat16* __restrict__ x_dec,
                 const __hip_bfloat16* __restrict__ pre_b,
                 float* __restrict__ out_aws, int t) {
  __shared__ float s_awp[TENC + KCONV - 1];
  __shared__ float s_cawp[TENC + KCONV - 1];
  __shared__ float s_q[ADIM];
  __shared__ float red[256];
  __shared__ float s_awn[TENC];

  const int b = blockIdx.x, tid = threadIdx.x;
  // halo zeros
  if (tid < 15) { s_awp[tid] = 0; s_cawp[tid] = 0; s_awp[271 + tid] = 0; s_cawp[271 + tid] = 0; }
  // async-stage aw/caw rows into LDS
  {
    uint32_t l0 = lds_off32(&s_awp[15 + tid]);
    unsigned long long a0 = (unsigned long long)(uintptr_t)(aw + b * TENC + tid);
    asm volatile("global_load_async_to_lds_b32 %0, %1, off" :: "v"(l0), "v"(a0) : "memory");
    uint32_t l1 = lds_off32(&s_cawp[15 + tid]);
    unsigned long long a1 = (unsigned long long)(uintptr_t)(caw + b * TENC + tid);
    asm volatile("global_load_async_to_lds_b32 %0, %1, off" :: "v"(l1), "v"(a1) : "memory");
    asm volatile("s_wait_asynccnt 0x0" ::: "memory");
  }
  if (tid < ADIM) {
    float s = score_b[tid];
    for (int h = 0; h < HDIM; ++h) s += h1[b * HDIM + h] * q_proj[h * ADIM + tid];
    s_q[tid] = s;
  }
  __syncthreads();

  // location conv features for position `tid`
  float convf[FDIM];
  #pragma unroll
  for (int f = 0; f < FDIM; ++f) {
    const float* w0 = loc_conv + (f * 2 + 0) * KCONV;
    const float* w1 = loc_conv + (f * 2 + 1) * KCONV;
    float s = 0;
    for (int j = 0; j < KCONV; ++j) s += s_awp[tid + j] * w0[j] + s_cawp[tid + j] * w1[j];
    convf[f] = s;
  }
  // score
  const float* krow = keys + ((size_t)b * TENC + tid) * ADIM;
  float sc = 0;
  for (int a = 0; a < ADIM; ++a) {
    float loc = 0;
    #pragma unroll
    for (int f = 0; f < FDIM; ++f) loc += convf[f] * loc_proj[f * ADIM + a];
    sc += tanhf(s_q[a] + krow[a] + loc) * score_v[a];
  }
  // softmax over 256 positions
  red[tid] = sc; __syncthreads();
  for (int s = 128; s > 0; s >>= 1) { if (tid < s) red[tid] = fmaxf(red[tid], red[tid + s]); __syncthreads(); }
  float mx = red[0]; __syncthreads();
  float ex = __expf(sc - mx);
  red[tid] = ex; __syncthreads();
  for (int s = 128; s > 0; s >>= 1) { if (tid < s) red[tid] += red[tid + s]; __syncthreads(); }
  float a_n = ex / red[0];

  aw[b * TENC + tid]  = a_n;
  caw[b * TENC + tid] = s_cawp[15 + tid] + a_n;
  out_aws[((size_t)b * TDEC + t) * TENC + tid] = a_n;
  s_awn[tid] = a_n;
  __syncthreads();

  // context + propagate into packed x vectors
  for (int e = tid; e < EDIM; e += 256) {
    float c = 0;
    for (int tt = 0; tt < TENC; ++tt) c += s_awn[tt] * enc[((size_t)b * TENC + tt) * EDIM + e];
    ctx[b * EDIM + e] = c;
    __hip_bfloat16 cb = __float2bfloat16(c);
    x_att[(size_t)b * XK_ATT + PDIM + e] = cb;
    x_dec[(size_t)b * XK_DEC + HDIM + e] = cb;
  }
  for (int h = tid; h < HDIM; h += 256) {
    __hip_bfloat16 v = __float2bfloat16(h1[b * HDIM + h]);
    x_att[(size_t)b * XK_ATT + PDIM + EDIM + h] = v;   // recurrent h1 for step t+1
    x_dec[(size_t)b * XK_DEC + h] = v;                 // h1 input for decoder LSTM now
  }
  if (t + 1 < TDEC)
    x_att[(size_t)b * XK_ATT + tid] = pre_b[((size_t)b * TDEC + t + 1) * PDIM + tid];
}

// feat / stop projections; write h2 bf16 into decoder recurrent slot.
__global__ void __launch_bounds__(128)
output_kernel(const float* __restrict__ h2, const float* __restrict__ ctx,
              const float* __restrict__ feat_W, const float* __restrict__ feat_b,
              const float* __restrict__ stop_W, const float* __restrict__ stop_b,
              __hip_bfloat16* __restrict__ x_dec,
              float* __restrict__ out_feats, float* __restrict__ out_stops, int t) {
  const int b = blockIdx.x, tid = threadIdx.x;
  if (tid < NMEL) {
    float s = feat_b[tid];
    for (int k = 0; k < HDIM; ++k) s += h2[b * HDIM + k] * feat_W[k * NMEL + tid];
    for (int k = 0; k < EDIM; ++k) s += ctx[b * EDIM + k] * feat_W[(HDIM + k) * NMEL + tid];
    out_feats[((size_t)b * TDEC + t) * NMEL + tid] = s;
  } else if (tid == NMEL) {
    float s = stop_b[0];
    for (int k = 0; k < HDIM; ++k) s += h2[b * HDIM + k] * stop_W[k];
    for (int k = 0; k < EDIM; ++k) s += ctx[b * EDIM + k] * stop_W[HDIM + k];
    out_stops[(size_t)b * TDEC + t] = s;
  }
  for (int h = tid; h < HDIM; h += 128)
    x_dec[(size_t)b * XK_DEC + HDIM + EDIM + h] = __float2bfloat16(h2[b * HDIM + h]);
}

// ---------------------------------------------------------------------------
extern "C" void kernel_launch(void* const* d_in, const int* in_sizes, int n_in,
                              void* d_out, int out_size, void* d_ws, size_t ws_size,
                              hipStream_t stream) {
  (void)in_sizes; (void)n_in; (void)out_size; (void)ws_size;
  const float* enc       = (const float*)d_in[0];
  const float* mel       = (const float*)d_in[1];
  const float* prenet_W1 = (const float*)d_in[2];
  const float* prenet_b1 = (const float*)d_in[3];
  const float* prenet_W2 = (const float*)d_in[4];
  const float* prenet_b2 = (const float*)d_in[5];
  const float* att_Wih   = (const float*)d_in[6];
  const float* att_Whh   = (const float*)d_in[7];
  const float* att_b     = (const float*)d_in[8];
  const float* dec_Wih   = (const float*)d_in[9];
  const float* dec_Whh   = (const float*)d_in[10];
  const float* dec_b     = (const float*)d_in[11];
  const float* q_proj    = (const float*)d_in[12];
  const float* v_proj    = (const float*)d_in[13];
  const float* loc_conv  = (const float*)d_in[14];
  const float* loc_proj  = (const float*)d_in[15];
  const float* score_b   = (const float*)d_in[16];
  const float* score_v   = (const float*)d_in[17];
  const float* feat_W    = (const float*)d_in[18];
  const float* feat_b    = (const float*)d_in[19];
  const float* stop_W    = (const float*)d_in[20];
  const float* stop_b    = (const float*)d_in[21];

  float* out = (float*)d_out;
  float* out_feats = out;
  float* out_stops = out + (size_t)B_ * TDEC * NMEL;
  float* out_aws   = out_stops + (size_t)B_ * TDEC;

  char* p = (char*)d_ws;
  auto carve = [&](size_t bytes) -> void* {
    void* r = (void*)p; p += (bytes + 255) & ~(size_t)255; return r;
  };
  uint32_t* W_att        = (uint32_t*)carve((size_t)XK_ATT * NGATE * 2);
  uint32_t* W_dec        = (uint32_t*)carve((size_t)XK_DEC * NGATE * 2);
  float* tmp1            = (float*)carve((size_t)B_ * TDEC * PDIM * 4);
  __hip_bfloat16* pre_b  = (__hip_bfloat16*)carve((size_t)B_ * TDEC * PDIM * 2);
  float* keysbuf         = (float*)carve((size_t)B_ * TENC * ADIM * 4);
  __hip_bfloat16* x_att  = (__hip_bfloat16*)carve((size_t)B_ * XK_ATT * 2);
  __hip_bfloat16* x_dec  = (__hip_bfloat16*)carve((size_t)B_ * XK_DEC * 2);
  float* h1  = (float*)carve((size_t)B_ * HDIM * 4);
  float* c1  = (float*)carve((size_t)B_ * HDIM * 4);
  float* h2  = (float*)carve((size_t)B_ * HDIM * 4);
  float* c2  = (float*)carve((size_t)B_ * HDIM * 4);
  float* aw  = (float*)carve((size_t)B_ * TENC * 4);
  float* caw = (float*)carve((size_t)B_ * TENC * 4);
  float* ctx = (float*)carve((size_t)B_ * EDIM * 4);

  // dynamic LDS sizes for the LSTM kernels (x padded + 16x512 gate area)
  auto lds_bytes = [](int xK) -> size_t {
    int ndw = (xK >> 1) * 16;
    int xdw = ndw + 4 * (ndw >> 8) + 16;
    xdw = (xdw + 3) & ~3;
    return (size_t)(xdw + 16 * 512) * 4;
  };
  size_t shm_att = lds_bytes(XK_ATT);   // ~91 KB
  size_t shm_dec = lds_bytes(XK_DEC);   // ~116 KB

  // one-time (per call) parallel pre-work
  convert_w_kernel<<<2048, 256, 0, stream>>>(att_Wih, att_Whh, XK_ATT - HDIM, XK_ATT / 32, W_att);
  convert_w_kernel<<<2048, 256, 0, stream>>>(dec_Wih, dec_Whh, XK_DEC - HDIM, XK_DEC / 32, W_dec);
  prenet1_kernel<<<1600, 256, 0, stream>>>(mel, prenet_W1, prenet_b1, tmp1);
  prenet2_kernel<<<1600, 256, 0, stream>>>(tmp1, prenet_W2, prenet_b2, pre_b);
  keys_kernel<<<2048, 256, 0, stream>>>(enc, v_proj, keysbuf);
  init_state_kernel<<<64, 256, 0, stream>>>(h1, c1, h2, c2, aw, caw, ctx, x_att, x_dec, pre_b);

  // sequential decode: 4 stream-ordered kernels per step
  for (int t = 0; t < TDEC; ++t) {
    lstm_step_kernel<<<32, 256, shm_att, stream>>>(W_att, att_b, (const uint32_t*)x_att, XK_ATT, c1, h1);
    attention_kernel<<<B_, 256, 0, stream>>>(enc, keysbuf, h1, q_proj, loc_conv, loc_proj,
                                             score_b, score_v, aw, caw, ctx,
                                             x_att, x_dec, pre_b, out_aws, t);
    lstm_step_kernel<<<32, 256, shm_dec, stream>>>(W_dec, dec_b, (const uint32_t*)x_dec, XK_DEC, c2, h2);
    output_kernel<<<B_, 128, 0, stream>>>(h2, ctx, feat_W, feat_b, stop_W, stop_b,
                                          x_dec, out_feats, out_stops, t);
  }
}
